// CrossScaleAttention_27350351741658
// MI455X (gfx1250) — compile-verified, matrix operand order checked by reading, not run on hardware
//
#include <hip/hip_runtime.h>

// ---------------- CDNA5 WMMA types ----------------
typedef __attribute__((ext_vector_type(16))) __bf16 v16bf;
typedef __attribute__((ext_vector_type(8)))  float  v8f;

union F16x16 { v16bf v; unsigned short u[16]; uint4 q[2]; };

__device__ __forceinline__ unsigned short f2bf(float f) {
  unsigned int u = __float_as_uint(f);
  u += 0x7FFFu + ((u >> 16) & 1u);           // round-to-nearest-even
  return (unsigned short)(u >> 16);
}

// Build a 16-bit A/B WMMA fragment from a row-major 32-element K window.
// Per ISA layout: lane-half `sel` holds K = [sel*8, sel*8+8) ++ [16+sel*8, 16+sel*8+8).
__device__ __forceinline__ v16bf ld_frag_bf16(const unsigned short* p, int sel) {
  F16x16 r;
  r.q[0] = *(const uint4*)(p + sel * 8);
  r.q[1] = *(const uint4*)(p + 16 + sel * 8);
  return r.v;
}

#define WMMA_BF16(A, B, C) \
  __builtin_amdgcn_wmma_f32_16x16x32_bf16(false, (A), false, (B), (short)0, (C), false, false)

// ---------------- constants ----------------
static constexpr int N_TOK = 256;
static constexpr int DIM   = 192;
static constexpr int HEADS = 6;
static constexpr int HD    = 32;
static constexpr int BWIN  = 512;
static constexpr int ROWS  = BWIN * N_TOK;                         // 131072
static constexpr size_t QKV_ELEMS = (size_t)BWIN * HEADS * N_TOK * HD; // 25165824
static constexpr size_t X_ELEMS   = (size_t)ROWS * DIM;                // 25165824
static constexpr size_t QKVW_ELEMS = (size_t)3 * DIM * DIM;            // 110592
static constexpr size_t PROJW_ELEMS = (size_t)DIM * DIM;               // 36864

// ------------------------------------------------------------------
// Prep kernel: bulk f32 -> bf16 (8 elements / thread, exact sizing)
// ------------------------------------------------------------------
__global__ __launch_bounds__(256) void wattn_cvt_bf16(
    const float* __restrict__ in, unsigned short* __restrict__ out) {
  const size_t i = ((size_t)blockIdx.x * 256 + threadIdx.x) * 8;
  const float4 a = *(const float4*)(in + i);
  const float4 b = *(const float4*)(in + i + 4);
  union { unsigned short u[8]; uint4 q; } r;
  r.u[0] = f2bf(a.x); r.u[1] = f2bf(a.y); r.u[2] = f2bf(a.z); r.u[3] = f2bf(a.w);
  r.u[4] = f2bf(b.x); r.u[5] = f2bf(b.y); r.u[6] = f2bf(b.z); r.u[7] = f2bf(b.w);
  *(uint4*)(out + i) = r.q;
}

// ------------------------------------------------------------------
// Prep kernel: expand relative-position bias to bias[h][n][m] (f32)
// ------------------------------------------------------------------
__global__ __launch_bounds__(256) void wattn_bias_expand(
    const float* __restrict__ bias_table, const int* __restrict__ rel_index,
    float* __restrict__ biasbuf) {
  const int idx = blockIdx.x * 256 + threadIdx.x;      // < 6*65536
  const int h = idx >> 16;
  const int r = idx & 65535;
  biasbuf[idx] = bias_table[rel_index[r] * HEADS + h];
}

// ------------------------------------------------------------------
// Kernel 1: QKV GEMM (131072 x 576 x 192), all-bf16 WMMA.
//   grid = (1024, 3); grid.y == section (0:Q scaled, 1:K, 2:V transposed).
//   Each wave: 16 rows x 192 cols = 12 C tiles, 72 WMMAs.
// ------------------------------------------------------------------
__global__ __launch_bounds__(256) void wattn_qkv_gemm(
    const unsigned short* __restrict__ xbf, const unsigned short* __restrict__ wbf,
    unsigned short* __restrict__ qbuf, unsigned short* __restrict__ kbuf,
    unsigned short* __restrict__ vtbuf) {
  const int lane  = threadIdx.x & 31;
  const int wave  = threadIdx.x >> 5;
  const int l16   = lane & 15;
  const int sel   = lane >> 4;
  const int rowBase = blockIdx.x * 128 + wave * 16;
  const int sec     = blockIdx.y;                  // 0=Q, 1=K, 2=V
  const int colBase = sec * DIM;

  const v8f vzero = {0.f, 0.f, 0.f, 0.f, 0.f, 0.f, 0.f, 0.f};
  v8f acc[12];
#pragma unroll
  for (int t = 0; t < 12; ++t) acc[t] = vzero;

  const unsigned short* arow = xbf + (size_t)(rowBase + l16) * DIM;
#pragma unroll
  for (int ks = 0; ks < 6; ++ks) {                 // K = 192, 32 per WMMA
    const v16bf af = ld_frag_bf16(arow + ks * 32, sel);
#pragma unroll
    for (int t = 0; t < 12; ++t) {
      const int col = colBase + t * 16 + l16;
      const v16bf bf = ld_frag_bf16(wbf + (size_t)col * DIM + ks * 32, sel);
      acc[t] = WMMA_BF16(af, bf, acc[t]);
    }
  }

  const float scale = 0.17677669529663687f;        // 32^-0.5
#pragma unroll
  for (int t = 0; t < 12; ++t) {
    const int f    = t * 16 + l16;                 // feature within section
    const int head = f >> 5;
    const int dd   = f & 31;
#pragma unroll
    for (int r = 0; r < 8; ++r) {
      const int row = rowBase + r + (sel ? 8 : 0);
      const int win = row >> 8;
      const int tok = row & 255;
      const float v = acc[t][r];
      const size_t wh = (size_t)win * HEADS + head;
      if (sec == 0)       qbuf [(wh * N_TOK + tok) * HD + dd]  = f2bf(v * scale);
      else if (sec == 1)  kbuf [(wh * N_TOK + tok) * HD + dd]  = f2bf(v);
      else                vtbuf[(wh * HD + dd) * N_TOK + tok]  = f2bf(v);
    }
  }
}

// ------------------------------------------------------------------
// Kernel 2: per-(window,head) attention, fully LDS-resident.
//   blocks = 512*6, 256 threads (8 waves). Dyn LDS = 240 KB (< 320 KB/WGP).
// ------------------------------------------------------------------
__global__ __launch_bounds__(256) void wattn_attention(
    const unsigned short* __restrict__ qbuf, const unsigned short* __restrict__ kbuf,
    const unsigned short* __restrict__ vtbuf, const float* __restrict__ biasbuf,
    unsigned short* __restrict__ abuf) {
  extern __shared__ unsigned char smem[];
  unsigned short* sQ  = (unsigned short*)smem;                  // 256x32 bf16 = 16 KB
  unsigned short* sK  = sQ + N_TOK * HD;                        // 16 KB
  unsigned short* sVt = sK + N_TOK * HD;                        // 32x256 bf16 = 16 KB
  float*          sSa = (float*)(smem + 3 * 16384);             // 8 x 16x256 f32 = 128 KB
  unsigned short* sPa = (unsigned short*)(smem + 3 * 16384 + 8 * 16 * N_TOK * 4); // 64 KB

  const int tid  = threadIdx.x;
  const int lane = tid & 31;
  const int wave = tid >> 5;
  const int l16  = lane & 15;
  const int sel  = lane >> 4;
  const int wh   = blockIdx.x;                 // win*6 + head
  const int h    = wh % HEADS;
  const int win  = wh / HEADS;
  const size_t base = (size_t)wh * (N_TOK * HD);

  { // stage Q, K, Vt -> LDS (1024 uint4 each)
    const uint4* gq = (const uint4*)(qbuf + base);
    const uint4* gk = (const uint4*)(kbuf + base);
    const uint4* gv = (const uint4*)(vtbuf + base);
    uint4* lq = (uint4*)sQ; uint4* lk = (uint4*)sK; uint4* lv = (uint4*)sVt;
#pragma unroll
    for (int i = 0; i < 4; ++i) {
      lq[tid + 256 * i] = gq[tid + 256 * i];
      lk[tid + 256 * i] = gk[tid + 256 * i];
      lv[tid + 256 * i] = gv[tid + 256 * i];
    }
  }
  __syncthreads();

  float*          sS = sSa + wave * (16 * N_TOK);
  unsigned short* sP = sPa + wave * (16 * N_TOK);
  const float* bias_h = biasbuf + h * (N_TOK * N_TOK);

  for (int it = 0; it < 2; ++it) {             // 8 waves x 2 strips = 256 rows
    const int row0 = (wave + it * 8) * 16;

    // ---- scores S = Q Kᵀ + bias (16 x 256 strip, 16 WMMAs) ----
    const v16bf aq = ld_frag_bf16(sQ + (row0 + l16) * HD, sel);
#pragma unroll
    for (int t = 0; t < 16; ++t) {
      const v16bf bk = ld_frag_bf16(sK + (t * 16 + l16) * HD, sel);
      v8f c = {0.f, 0.f, 0.f, 0.f, 0.f, 0.f, 0.f, 0.f};
      c = WMMA_BF16(aq, bk, c);
#pragma unroll
      for (int r = 0; r < 8; ++r) {
        const int M = r + (sel ? 8 : 0);
        const int n = row0 + M;
        const int m = t * 16 + l16;
        sS[M * N_TOK + m] = c[r] + bias_h[n * N_TOK + m];
      }
    }
    __syncthreads();

    // ---- softmax over m: 2 lanes per row, halves combined via shfl_xor(16) ----
    {
      float* rowp = sS + l16 * N_TOK + sel * 128;
      float mx = -3.4e38f;
#pragma unroll 8
      for (int j = 0; j < 32; ++j) {
        const float4 v = ((const float4*)rowp)[j];
        mx = fmaxf(mx, fmaxf(fmaxf(v.x, v.y), fmaxf(v.z, v.w)));
      }
      mx = fmaxf(mx, __shfl_xor(mx, 16, 32));
      float sum = 0.f;
#pragma unroll 8
      for (int j = 0; j < 32; ++j) {
        float4 v = ((float4*)rowp)[j];
        v.x = __expf(v.x - mx); v.y = __expf(v.y - mx);
        v.z = __expf(v.z - mx); v.w = __expf(v.w - mx);
        ((float4*)rowp)[j] = v;
        sum += v.x + v.y + v.z + v.w;
      }
      sum += __shfl_xor(sum, 16, 32);
      const float inv = 1.0f / sum;
      unsigned short* prow = sP + l16 * N_TOK + sel * 128;
#pragma unroll 8
      for (int j = 0; j < 32; ++j) {
        const float4 v = ((const float4*)rowp)[j];
        ushort4 p;
        p.x = f2bf(v.x * inv); p.y = f2bf(v.y * inv);
        p.z = f2bf(v.z * inv); p.w = f2bf(v.w * inv);
        ((ushort4*)prow)[j] = p;
      }
    }
    __syncthreads();

    // ---- O = P V  (16 x 32 strip; K = 256 -> 8 WMMAs per 16-col tile) ----
#pragma unroll
    for (int tc = 0; tc < 2; ++tc) {
      v8f o = {0.f, 0.f, 0.f, 0.f, 0.f, 0.f, 0.f, 0.f};
#pragma unroll
      for (int kt = 0; kt < 8; ++kt) {
        const v16bf ap = ld_frag_bf16(sP + l16 * N_TOK + kt * 32, sel);
        const v16bf bv = ld_frag_bf16(sVt + (tc * 16 + l16) * N_TOK + kt * 32, sel);
        o = WMMA_BF16(ap, bv, o);
      }
#pragma unroll
      for (int r = 0; r < 8; ++r) {
        const int M   = r + (sel ? 8 : 0);
        const int tok = row0 + M;
        const int d   = tc * 16 + l16;
        abuf[((size_t)win * N_TOK + tok) * DIM + h * HD + d] = f2bf(o[r]);
      }
    }
    __syncthreads();
  }
}

// ------------------------------------------------------------------
// Kernel 3: output projection (131072 x 192 x 192) + bias -> f32 out.
//   Each wave: 16 rows x 192 cols = 12 C tiles; A read exactly once.
// ------------------------------------------------------------------
__global__ __launch_bounds__(256) void wattn_proj_gemm(
    const unsigned short* __restrict__ abuf, const unsigned short* __restrict__ pwbf,
    const float* __restrict__ b, float* __restrict__ out) {
  const int lane  = threadIdx.x & 31;
  const int wave  = threadIdx.x >> 5;
  const int l16   = lane & 15;
  const int sel   = lane >> 4;
  const int rowBase = blockIdx.x * 128 + wave * 16;

  const v8f vzero = {0.f, 0.f, 0.f, 0.f, 0.f, 0.f, 0.f, 0.f};
  v8f acc[12];
#pragma unroll
  for (int t = 0; t < 12; ++t) acc[t] = vzero;

  const unsigned short* arow = abuf + (size_t)(rowBase + l16) * DIM;
#pragma unroll
  for (int ks = 0; ks < 6; ++ks) {
    const v16bf af = ld_frag_bf16(arow + ks * 32, sel);
#pragma unroll
    for (int t = 0; t < 12; ++t) {
      const int col = t * 16 + l16;
      const v16bf bf = ld_frag_bf16(pwbf + (size_t)col * DIM + ks * 32, sel);
      acc[t] = WMMA_BF16(af, bf, acc[t]);
    }
  }

#pragma unroll
  for (int t = 0; t < 12; ++t) {
    const int col = t * 16 + l16;
    const float bc = b[col];
#pragma unroll
    for (int r = 0; r < 8; ++r) {
      const int row = rowBase + r + (sel ? 8 : 0);
      out[(size_t)row * DIM + col] = acc[t][r] + bc;
    }
  }
}

// ------------------------------------------------------------------
extern "C" void kernel_launch(void* const* d_in, const int* in_sizes, int n_in,
                              void* d_out, int out_size, void* d_ws, size_t ws_size,
                              hipStream_t stream) {
  const float* x          = (const float*)d_in[0];
  const float* qkv_w      = (const float*)d_in[1];
  const float* proj_w     = (const float*)d_in[2];
  const float* proj_b     = (const float*)d_in[3];
  const float* bias_table = (const float*)d_in[4];
  const int*   rel_index  = (const int*)d_in[5];
  float* out = (float*)d_out;

  unsigned short* qbuf  = (unsigned short*)d_ws;
  unsigned short* kbuf  = qbuf  + QKV_ELEMS;
  unsigned short* vtbuf = kbuf  + QKV_ELEMS;
  unsigned short* abuf  = vtbuf + QKV_ELEMS;
  unsigned short* xbf   = abuf  + QKV_ELEMS;
  unsigned short* wbf   = xbf   + X_ELEMS;
  unsigned short* pwbf  = wbf   + QKVW_ELEMS;
  float*          biasbuf = (float*)(pwbf + PROJW_ELEMS + 8 /*align*/);

  // ---- prep: bf16 conversions + bias expansion (memory-bound, one-shot) ----
  wattn_cvt_bf16<<<dim3(X_ELEMS / 2048),     dim3(256), 0, stream>>>(x, xbf);
  wattn_cvt_bf16<<<dim3(QKVW_ELEMS / 2048),  dim3(256), 0, stream>>>(qkv_w, wbf);
  wattn_cvt_bf16<<<dim3(PROJW_ELEMS / 2048), dim3(256), 0, stream>>>(proj_w, pwbf);
  wattn_bias_expand<<<dim3((HEADS * 65536) / 256), dim3(256), 0, stream>>>(
      bias_table, rel_index, biasbuf);

  // ---- main pipeline ----
  wattn_qkv_gemm<<<dim3(ROWS / 128, 3), dim3(256), 0, stream>>>(
      xbf, wbf, qbuf, kbuf, vtbuf);

  const size_t attn_lds = 3 * 16384 + 8 * 16 * N_TOK * 4 + 8 * 16 * N_TOK * 2; // 245760
  wattn_attention<<<dim3(BWIN * HEADS), dim3(256), attn_lds, stream>>>(
      qbuf, kbuf, vtbuf, biasbuf, abuf);

  wattn_proj_gemm<<<dim3(ROWS / 128), dim3(256), 0, stream>>>(
      abuf, pwbf, proj_b, out);
}
